// RnnSpeaker_5102421147798
// MI455X (gfx1250) — matrix operand order repaired and checked by reading, not compiled, not run on hardware
//
#include <hip/hip_runtime.h>

// ---------------------------------------------------------------------------
// RnnSpeaker for gfx1250 (CDNA5 / MI455X), wave32 + v_wmma_f32_16x16x32_f16.
// Round 3: vocab GEMM now stages the shared B panel into LDS with the CDNA5
// async copy engine (global_load_async_to_lds_b128, ASYNCcnt, double buffer);
// GRU GEMMs keep the proven branch-free direct-load WMMA path.
// ---------------------------------------------------------------------------
#define D_EMB   256
#define D_HID   512
#define VOCAB   10000
#define SEQ_LEN 16
#define BATCH   1024
#define G3      (3 * D_HID)   // 1536
#define BOS     2
#define EOS     3

// LDS B-panel: 64 rows x 32 K f16, padded row stride 40 elems (80B) so rows
// 0..15 hit distinct banks; two buffers for double buffering.
#define BROW_PAD 40
#define BPANEL   (64 * BROW_PAD)

typedef __attribute__((ext_vector_type(16))) _Float16 v16h;
typedef __attribute__((ext_vector_type(8)))  float    v8f;
typedef __attribute__((ext_vector_type(4)))  float    v4f;

union HFrag { v16h v; uint4 q[2]; };

__device__ __forceinline__ v8f wmma16x16x32(v16h a, v16h b, v8f c) {
    return __builtin_amdgcn_wmma_f32_16x16x32_f16(false, a, false, b, (short)0, c, false, false);
}

// -------------------- CDNA5 async copy helpers -----------------------------
// ISA 15.18.3 op 98: each lane copies 16B from global memory straight into LDS;
// the instruction offset is added to BOTH addresses, tracked by ASYNCcnt.
__device__ __forceinline__ void async_b128(unsigned lds_off, const void* gsrc) {
    asm volatile("global_load_async_to_lds_b128 %0, %1, off"
                 :: "v"(lds_off), "v"(gsrc) : "memory");
}
__device__ __forceinline__ void async_b128_o16(unsigned lds_off, const void* gsrc) {
    asm volatile("global_load_async_to_lds_b128 %0, %1, off offset:16"
                 :: "v"(lds_off), "v"(gsrc) : "memory");
}
template <int NWAIT>
__device__ __forceinline__ void wait_asynccnt() {
#if __has_builtin(__builtin_amdgcn_s_wait_asynccnt)
    __builtin_amdgcn_s_wait_asynccnt(NWAIT);
#else
    asm volatile("s_wait_asynccnt %0" :: "i"(NWAIT) : "memory");
#endif
}

// Stage one 64x32 f16 B panel (rows nb..nb+63, K kc..kc+31) into LDS buffer.
// 128 threads x 32B each = 4KB; each thread issues 2 async b128 copies.
__device__ __forceinline__ void stage_b_panel(const _Float16* __restrict__ W, int K, int N,
                                              int nb, int kc, _Float16* sbuf, int tid) {
    int row  = tid >> 1;                     // 0..63
    int half = tid & 1;                      // which 32B of the 64B row slice
    int grow = nb + row; if (grow >= N) grow = N - 1;   // clamp edge block
    const _Float16* src = W + (size_t)grow * K + kc + half * 16;
    unsigned dst = (unsigned)(size_t)(sbuf + row * BROW_PAD + half * 16);
    async_b128(dst, src);
    async_b128_o16(dst, src);
}

// ---------------------------------------------------------------------------
// WMMA accumulate core (direct-load variant): wave computes 32(M) x 64(N)
// = 2x4 16x16 tiles. A f16 [M,K] row-major; W f16 [N,K] row-major (C=A*W^T).
// A frag (16x32): lane&15 = row; lanes 0-15 hold K 0-7 & 16-23, lanes 16-31
//                 hold K 8-15 & 24-31 (ISA 7.12.2).
// B frag (32x16): lane&15 = col; same K split as A (0-7 & 16-23 / 8-15 & 24-31).
// nt[4] clamped tile-column bases -> fully branch-free k-loop.
// ---------------------------------------------------------------------------
__device__ __forceinline__ void mma_core(const _Float16* __restrict__ A,
                                         const _Float16* __restrict__ W,
                                         int K, int m0, const int* nt, int lane,
                                         v8f (&acc)[8]) {
    const int rowA  = lane & 15;
    const int koffA = (lane & 16) ? 8 : 0;
    const int koffB = (lane & 16) ? 8 : 0;
    const _Float16* pa0 = A + (size_t)(m0 + rowA) * K + koffA;
    const _Float16* pa1 = pa0 + (size_t)16 * K;
    const _Float16* pb0 = W + (size_t)(nt[0] + rowA) * K + koffB;
    const _Float16* pb1 = W + (size_t)(nt[1] + rowA) * K + koffB;
    const _Float16* pb2 = W + (size_t)(nt[2] + rowA) * K + koffB;
    const _Float16* pb3 = W + (size_t)(nt[3] + rowA) * K + koffB;

    for (int k = 0; k < K; k += 32) {
        HFrag a0, a1, b0, b1, b2, b3;
        a0.q[0] = *(const uint4*)(pa0 + k);  a0.q[1] = *(const uint4*)(pa0 + k + 16);
        a1.q[0] = *(const uint4*)(pa1 + k);  a1.q[1] = *(const uint4*)(pa1 + k + 16);
        b0.q[0] = *(const uint4*)(pb0 + k);  b0.q[1] = *(const uint4*)(pb0 + k + 16);
        b1.q[0] = *(const uint4*)(pb1 + k);  b1.q[1] = *(const uint4*)(pb1 + k + 16);
        b2.q[0] = *(const uint4*)(pb2 + k);  b2.q[1] = *(const uint4*)(pb2 + k + 16);
        b3.q[0] = *(const uint4*)(pb3 + k);  b3.q[1] = *(const uint4*)(pb3 + k + 16);
        acc[0] = wmma16x16x32(a0.v, b0.v, acc[0]);
        acc[1] = wmma16x16x32(a1.v, b0.v, acc[1]);
        acc[2] = wmma16x16x32(a0.v, b1.v, acc[2]);
        acc[3] = wmma16x16x32(a1.v, b1.v, acc[3]);
        acc[4] = wmma16x16x32(a0.v, b2.v, acc[4]);
        acc[5] = wmma16x16x32(a1.v, b2.v, acc[5]);
        acc[6] = wmma16x16x32(a0.v, b3.v, acc[6]);
        acc[7] = wmma16x16x32(a1.v, b3.v, acc[7]);
    }
}

// -------------------- fp32 -> fp16 conversion ------------------------------
__global__ void f32_to_f16_kernel(const float* __restrict__ in, _Float16* __restrict__ out, int n) {
    for (int i = blockIdx.x * blockDim.x + threadIdx.x; i < n; i += gridDim.x * blockDim.x)
        out[i] = (_Float16)in[i];
}

// -------------------- generic WMMA GEMM: C = A * W^T + bias ----------------
// Block = 4 waves stacked in M: block tile 128(M) x 64(N).
__global__ void __launch_bounds__(128)
gemm_bias_kernel(const _Float16* __restrict__ A, const _Float16* __restrict__ W,
                 const float* __restrict__ bias, float* __restrict__ C,
                 int M, int N, int K) {
    const int lane = threadIdx.x & 31;
    const int wave = threadIdx.x >> 5;
    const int m0 = blockIdx.y * 128 + wave * 32;
    const int nb = blockIdx.x * 64;
    if (m0 >= M) return;                         // uniform per wave

    int  nt[4];
    bool valid[4];
    #pragma unroll
    for (int j = 0; j < 4; ++j) {
        int n = nb + 16 * j;
        valid[j] = (n + 16) <= N;                // N is a multiple of 16
        nt[j]    = valid[j] ? n : 0;             // clamp -> branch-free loop
    }

    v8f acc[8] = {};
    mma_core(A, W, K, m0, nt, lane, acc);

    // C/D layout: lane&15 = N, VGPR r = M {r | r+8 by lane half}
    const int nl    = lane & 15;
    const int mhalf = (lane & 16) ? 8 : 0;
    #pragma unroll
    for (int j = 0; j < 4; ++j) {
        if (!valid[j]) continue;
        float bv = bias[nt[j] + nl];
        #pragma unroll
        for (int r = 0; r < 8; ++r) {
            C[(size_t)(m0      + mhalf + r) * N + nt[j] + nl] = acc[2 * j    ][r] + bv;
            C[(size_t)(m0 + 16 + mhalf + r) * N + nt[j] + nl] = acc[2 * j + 1][r] + bv;
        }
    }
}

// -------------------- gumbel noise (counter-based hash) --------------------
__device__ __forceinline__ float gumbel_noise(int step, int m, int v) {
    unsigned x = (unsigned)(step * 0x9E3779B9u) ^ (unsigned)(m * 0x85EBCA77u)
               ^ (unsigned)(v * 0xC2B2AE3Du);
    x ^= x >> 16; x *= 0x7FEB352Du; x ^= x >> 15; x *= 0x846CA68Bu; x ^= x >> 16;
    float u = ((x >> 8) + 0.5f) * (1.0f / 16777216.0f);   // (0,1)
    return -__logf(-__logf(u));
}

// -------------------- vocab GEMM fused with gumbel + argmax ----------------
// B panel async-staged into LDS (double buffered), shared by all 4 waves.
__global__ void __launch_bounds__(128)
vocab_argmax_kernel(const _Float16* __restrict__ A, const _Float16* __restrict__ W,
                    const float* __restrict__ bias, unsigned long long* __restrict__ amax,
                    int M, int N, int K, int step) {
    __shared__ _Float16 sB[2 * BPANEL];

    const int tid  = threadIdx.x;
    const int lane = tid & 31;
    const int wave = tid >> 5;
    const int m0 = blockIdx.y * 128 + wave * 32;
    const int nb = blockIdx.x * 64;

    int  nt[4];
    bool valid[4];
    #pragma unroll
    for (int j = 0; j < 4; ++j) {
        int n = nb + 16 * j;
        valid[j] = (n + 16) <= N;
        nt[j]    = valid[j] ? n : 0;
    }

    const int rowA  = lane & 15;
    const int koffA = (lane & 16) ? 8 : 0;
    const int koffB = (lane & 16) ? 8 : 0;
    const _Float16* pa0 = A + (size_t)(m0 + rowA) * K + koffA;
    const _Float16* pa1 = pa0 + (size_t)16 * K;

    v8f acc[8] = {};

    stage_b_panel(W, K, N, nb, 0, sB, tid);          // prologue: fill buffer 0
    for (int kc = 0; kc < K; kc += 32) {
        const int  cb   = (kc >> 5) & 1;
        const bool more = (kc + 32) < K;
        if (more) stage_b_panel(W, K, N, nb, kc + 32, sB + (cb ^ 1) * BPANEL, tid);
        if (more) wait_asynccnt<2>(); else wait_asynccnt<0>();
        __syncthreads();                              // current panel visible to all

        const _Float16* sbuf = sB + cb * BPANEL;
        HFrag a0, a1;
        a0.q[0] = *(const uint4*)(pa0 + kc);  a0.q[1] = *(const uint4*)(pa0 + kc + 16);
        a1.q[0] = *(const uint4*)(pa1 + kc);  a1.q[1] = *(const uint4*)(pa1 + kc + 16);
        HFrag b[4];
        #pragma unroll
        for (int j = 0; j < 4; ++j) {
            const _Float16* sp = sbuf + (16 * j + rowA) * BROW_PAD + koffB;
            b[j].q[0] = *(const uint4*)(sp);
            b[j].q[1] = *(const uint4*)(sp + 16);
        }
        acc[0] = wmma16x16x32(a0.v, b[0].v, acc[0]);
        acc[1] = wmma16x16x32(a1.v, b[0].v, acc[1]);
        acc[2] = wmma16x16x32(a0.v, b[1].v, acc[2]);
        acc[3] = wmma16x16x32(a1.v, b[1].v, acc[3]);
        acc[4] = wmma16x16x32(a0.v, b[2].v, acc[4]);
        acc[5] = wmma16x16x32(a1.v, b[2].v, acc[5]);
        acc[6] = wmma16x16x32(a0.v, b[3].v, acc[6]);
        acc[7] = wmma16x16x32(a1.v, b[3].v, acc[7]);
        __syncthreads();                              // protect panel vs overwrite
    }

    const int nl    = lane & 15;
    const int mhalf = (lane & 16) ? 8 : 0;
    float bj[4];
    #pragma unroll
    for (int j = 0; j < 4; ++j) bj[j] = valid[j] ? bias[nt[j] + nl] : 0.0f;

    #pragma unroll
    for (int mt = 0; mt < 2; ++mt) {
        #pragma unroll
        for (int r = 0; r < 8; ++r) {
            const int m = m0 + mt * 16 + mhalf + r;
            float best = -3.0e38f;
            int   bidx = 0;
            #pragma unroll
            for (int j = 0; j < 4; ++j) {
                if (valid[j]) {
                    int   n = nt[j] + nl;
                    float v = acc[2 * j + mt][r] + bj[j] + gumbel_noise(step, m, n);
                    if (v > best || (v == best && n < bidx)) { best = v; bidx = n; }
                }
            }
            // reduce across the 16 lanes of this half (xor offsets stay in-half)
            #pragma unroll
            for (int off = 1; off < 16; off <<= 1) {
                float ov = __shfl_xor(best, off, 32);
                int   oi = __shfl_xor(bidx, off, 32);
                if (ov > best || (ov == best && oi < bidx)) { best = ov; bidx = oi; }
            }
            if (nl == 0) {
                unsigned kb = __float_as_uint(best);
                kb = (kb & 0x80000000u) ? ~kb : (kb | 0x80000000u);   // monotone map
                atomicMax(amax + m, ((unsigned long long)kb << 32) | (unsigned)bidx);
            }
        }
    }
}

// -------------------- per-step helpers -------------------------------------
__global__ void zero_amax_kernel(unsigned long long* amax) {
    int i = blockIdx.x * blockDim.x + threadIdx.x;
    if (i < BATCH) amax[i] = 0ULL;
}

// label = low 32 bits of packed argmax key; gather its embedding row (f16).
__global__ void __launch_bounds__(32)
extract_gather_kernel(const unsigned long long* __restrict__ amax,
                      const _Float16* __restrict__ emb_h,
                      int* __restrict__ labels_raw, _Float16* __restrict__ inp_h,
                      int t, int fixed_label) {
    const int b = blockIdx.x, lane = threadIdx.x;
    int lab = fixed_label;
    if (fixed_label < 0) {
        unsigned long long key = amax[b];
        lab = (int)(unsigned)(key & 0xFFFFFFFFULL);
        if (lane == 0) labels_raw[b * SEQ_LEN + t] = lab;
    }
    const uint4* src = (const uint4*)(emb_h + (size_t)lab * D_EMB);
    uint4*       dst = (uint4*)(inp_h + (size_t)b * D_EMB);
    dst[lane] = src[lane];   // 32 lanes x 16B = 256 f16
}

// GRU gate combine: gates ordered (r,z,n) as in torch GRUCell.
__global__ void gru_gate_kernel(const float* __restrict__ gi, const float* __restrict__ gh,
                                const float* __restrict__ hprev,
                                float* __restrict__ hout, _Float16* __restrict__ hout_h) {
    int i = blockIdx.x * blockDim.x + threadIdx.x;   // BATCH*D_HID
    int b = i >> 9, h = i & (D_HID - 1);
    const float* gib = gi + (size_t)b * G3;
    const float* ghb = gh + (size_t)b * G3;
    float r = 1.0f / (1.0f + __expf(-(gib[h]         + ghb[h])));
    float z = 1.0f / (1.0f + __expf(-(gib[D_HID + h] + ghb[D_HID + h])));
    float n = tanhf(gib[2 * D_HID + h] + r * ghb[2 * D_HID + h]);
    float o = (1.0f - z) * n + z * hprev[i];
    hout[i]   = o;
    hout_h[i] = (_Float16)o;
}

// -------------------- finalize: EOS forcing / pad masking ------------------
__global__ void finalize_labels_kernel(const int* __restrict__ labels_raw,
                                       int* __restrict__ first_eos,
                                       float* __restrict__ out_labels,
                                       float* __restrict__ out_caplen) {
    int b = blockIdx.x * blockDim.x + threadIdx.x;
    if (b >= BATCH) return;
    int fe = SEQ_LEN - 1;                      // labels[:, -1] forced to EOS
    for (int t = 0; t < SEQ_LEN - 1; ++t) {
        if (labels_raw[b * SEQ_LEN + t] == EOS) { fe = t; break; }
    }
    first_eos[b] = fe;
    for (int t = 0; t < SEQ_LEN; ++t) {
        int lab = (t == SEQ_LEN - 1) ? EOS : labels_raw[b * SEQ_LEN + t];
        out_labels[b * SEQ_LEN + t] = (t < fe) ? (float)lab : 0.0f;
    }
    out_caplen[b] = (float)(fe + 1);
}

// one block per (b,t) row: zero-fill 10000 floats (NT stores), set the hot 1.
__global__ void __launch_bounds__(256)
write_logits_kernel(const int* __restrict__ labels_raw, const int* __restrict__ first_eos,
                    float* __restrict__ out) {
    const int row = blockIdx.x;                 // BATCH*SEQ_LEN
    const int b = row >> 4, t = row & (SEQ_LEN - 1);
    const int fe  = first_eos[b];
    const int hot = (t < fe) ? labels_raw[b * SEQ_LEN + t] : 0;   // pad rows -> one_hot(0)
    float* base = out + (size_t)row * VOCAB;
    v4f z = {0.0f, 0.0f, 0.0f, 0.0f};
    v4f* p = (v4f*)base;
    for (int i = threadIdx.x; i < VOCAB / 4; i += 256)
        __builtin_nontemporal_store(z, p + i);
    __syncthreads();
    if (threadIdx.x == 0) base[hot] = 1.0f;
}

// ---------------------------------------------------------------------------
extern "C" void kernel_launch(void* const* d_in, const int* in_sizes, int n_in,
                              void* d_out, int out_size, void* d_ws, size_t ws_size,
                              hipStream_t stream) {
    (void)in_sizes; (void)n_in; (void)out_size; (void)ws_size;
    const float* h_img = (const float*)d_in[0];
    // d_in[1]=caps_in, d_in[2]=caps_in_lens, d_in[3]=sample_how : unused (gumbel path)
    const float* emb_w = (const float*)d_in[4];
    const float* w_ih  = (const float*)d_in[5];
    const float* w_hh  = (const float*)d_in[6];
    const float* b_ih  = (const float*)d_in[7];
    const float* b_hh  = (const float*)d_in[8];
    const float* w_voc = (const float*)d_in[9];
    const float* b_voc = (const float*)d_in[10];

    char* ws = (char*)d_ws;
    _Float16* emb_h   = (_Float16*)(ws + 0);          // 10000*256  f16
    _Float16* w_ih_h  = (_Float16*)(ws + 5120000);    // 1536*256   f16
    _Float16* w_hh_h  = (_Float16*)(ws + 5906432);    // 1536*512   f16
    _Float16* w_voc_h = (_Float16*)(ws + 7479296);    // 10000*512  f16
    float*    hid     = (float*)   (ws + 17719296);   // 1024*512   f32
    _Float16* hid_h   = (_Float16*)(ws + 19816448);   // 1024*512   f16
    _Float16* inp_h   = (_Float16*)(ws + 20865024);   // 1024*256   f16
    float*    gi      = (float*)   (ws + 21389312);   // 1024*1536  f32
    float*    gh      = (float*)   (ws + 27680768);   // 1024*1536  f32
    unsigned long long* amax = (unsigned long long*)(ws + 33972224); // 1024 u64
    int*      labels_raw = (int*)(ws + 33980416);     // 1024*16
    int*      first_eos  = (int*)(ws + 34045952);     // 1024

    float* out_logits = (float*)d_out;                                    // [B,T,V]
    float* out_labels = out_logits + (size_t)BATCH * SEQ_LEN * VOCAB;     // [B,T]
    float* out_caplen = out_labels + (size_t)BATCH * SEQ_LEN;             // [B]

    // ---- one-time weight conversion to f16 (resident in 192MB L2 after) ----
    f32_to_f16_kernel<<<2048, 256, 0, stream>>>(emb_w, emb_h,   VOCAB * D_EMB);
    f32_to_f16_kernel<<<1024, 256, 0, stream>>>(w_ih,  w_ih_h,  G3 * D_EMB);
    f32_to_f16_kernel<<<1024, 256, 0, stream>>>(w_hh,  w_hh_h,  G3 * D_HID);
    f32_to_f16_kernel<<<2048, 256, 0, stream>>>(w_voc, w_voc_h, VOCAB * D_HID);
    f32_to_f16_kernel<<<1024, 256, 0, stream>>>(h_img, hid_h,   BATCH * D_HID);

    dim3 blk(128);
    dim3 grid_gru(G3 / 64, BATCH / 128);                      // 24 x 8
    dim3 grid_voc((VOCAB + 63) / 64, BATCH / 128);            // 157 x 8

    // ---- initial step: embed BOS, one GRU step from h_img ----
    extract_gather_kernel<<<BATCH, 32, 0, stream>>>(amax, emb_h, labels_raw, inp_h, 0, BOS);
    gemm_bias_kernel<<<grid_gru, blk, 0, stream>>>(inp_h, w_ih_h, b_ih, gi, BATCH, G3, D_EMB);
    gemm_bias_kernel<<<grid_gru, blk, 0, stream>>>(hid_h, w_hh_h, b_hh, gh, BATCH, G3, D_HID);
    gru_gate_kernel<<<(BATCH * D_HID) / 256, 256, 0, stream>>>(gi, gh, h_img, hid, hid_h);

    // ---- 16 sampling steps ----
    for (int t = 0; t < SEQ_LEN; ++t) {
        zero_amax_kernel<<<4, 256, 0, stream>>>(amax);
        vocab_argmax_kernel<<<grid_voc, blk, 0, stream>>>(hid_h, w_voc_h, b_voc, amax,
                                                          BATCH, VOCAB, D_HID, t);
        extract_gather_kernel<<<BATCH, 32, 0, stream>>>(amax, emb_h, labels_raw, inp_h, t, -1);
        gemm_bias_kernel<<<grid_gru, blk, 0, stream>>>(inp_h, w_ih_h, b_ih, gi, BATCH, G3, D_EMB);
        gemm_bias_kernel<<<grid_gru, blk, 0, stream>>>(hid_h, w_hh_h, b_hh, gh, BATCH, G3, D_HID);
        gru_gate_kernel<<<(BATCH * D_HID) / 256, 256, 0, stream>>>(gi, gh, hid, hid, hid_h);
    }

    // ---- finalize outputs ----
    finalize_labels_kernel<<<4, 256, 0, stream>>>(labels_raw, first_eos, out_labels, out_caplen);
    write_logits_kernel<<<BATCH * SEQ_LEN, 256, 0, stream>>>(labels_raw, first_eos, out_logits);
}
